// FViM_18047452577906
// MI455X (gfx1250) — matrix operand order repaired and checked by reading, not compiled, not run on hardware
//
#include <hip/hip_runtime.h>

typedef _Float16 f16;
typedef __attribute__((ext_vector_type(16))) _Float16 v16h;
typedef __attribute__((ext_vector_type(8)))  float    v8f;

#define DEVI __device__ __forceinline__
#define PI_F 3.14159265358979f

DEVI float geluf(float x){ return 0.5f*x*(1.f + erff(x*0.70710678118f)); }
DEVI float siluf(float x){ return x/(1.f + __expf(-x)); }
DEVI float softplusf(float x){ return (x > 20.f) ? x : log1pf(__expf(x)); }
DEVI float sshrink(float x){ float a = fabsf(x) - 0.01f; a = a > 0.f ? a : 0.f; return x >= 0.f ? a : -a; }

// wave32 butterfly reduction via ds_swizzle (xor masks 16,8,4,2,1)
DEVI float wave_sum(float v){
  v += __int_as_float(__builtin_amdgcn_ds_swizzle(__float_as_int(v), 0x401F));
  v += __int_as_float(__builtin_amdgcn_ds_swizzle(__float_as_int(v), 0x201F));
  v += __int_as_float(__builtin_amdgcn_ds_swizzle(__float_as_int(v), 0x101F));
  v += __int_as_float(__builtin_amdgcn_ds_swizzle(__float_as_int(v), 0x081F));
  v += __int_as_float(__builtin_amdgcn_ds_swizzle(__float_as_int(v), 0x041F));
  return v;
}

static inline int cdiv_h(long a, long b){ return (int)((a + b - 1) / b); }

DEVI void async_ld_b128(unsigned lds_addr, const float* gptr){
  asm volatile("global_load_async_to_lds_b128 %0, %1, off"
               :: "v"(lds_addr), "v"((unsigned long long)(size_t)gptr)
               : "memory");
}
DEVI void wait_async0(){ asm volatile("s_wait_asynccnt 0x0" ::: "memory"); }
DEVI void wait_async8(){ asm volatile("s_wait_asynccnt 0x8" ::: "memory"); }

#define LDS_RS 36   // padded LDS row stride (floats): 36r mod 64 distinct for r<16
                    // (conflict-free across lanes) and 144B keeps b128 16B-aligned

// stage one 16-row x 32-col fp32 panel (row-major, leading dim ld) into LDS
DEVI void stage_panel(const float* base, int r0, int ld, int k0, int lane, unsigned lds)
{
#pragma unroll
  for (int i = 0; i < 4; ++i) {
    int idx = i*128 + lane*4;
    int r = idx >> 5, cc = idx & 31;
    async_ld_b128(lds + (unsigned)(r*LDS_RS + cc)*4u,
                  base + (long)(r0 + r)*ld + (k0 + cc));
  }
}

// ---------------------------------------------------------------------------
// FAST WMMA GEMM (TRANSB only; M%16==0, N%16==0, K%32==0):
// 8 waves/block, one 16x16 C tile per wave. Each wave owns two private LDS
// slice pairs and software-pipelines: issue async b128 staging for step k+1,
// s_wait_asynccnt<=8 (step k landed, async loads complete in order), gather
// ISA-7.12.2 f16 fragments from LDS, v_wmma_f32_16x16x32_f16.
// ---------------------------------------------------------------------------
template<int ACT, int HASRES>
__global__ void __launch_bounds__(256)
wmma_gemm_fast_kernel(
    const float* __restrict__ A, const float* __restrict__ Bm, float* __restrict__ C,
    const float* __restrict__ bias, const float* __restrict__ scale,
    const float* __restrict__ shift, const float* __restrict__ res,
    int M, int N, int K, int lda, int ldb, int ldc,
    long sA, long sB, long sC, long sR, int amod, float sfac)
{
  __shared__ float shA[8][2][16*LDS_RS];
  __shared__ float shB[8][2][16*LDS_RS];
  int wv  = threadIdx.x >> 5;
  int wid = blockIdx.x * 8 + wv;
  int tnc = N >> 4;
  int tiles = (M >> 4) * tnc;
  if (wid >= tiles) return;                  // wave-uniform
  int bz = blockIdx.z;
  const float* Ab = A + (amod > 0 ? (long)(bz % amod) : (long)bz) * sA;
  const float* Bb = Bm + (long)bz * sB;
  float* Cb = C + (long)bz * sC;
  int tm = wid / tnc, tn = wid % tnc;
  int m0 = tm << 4, n0 = tn << 4;
  int lane = threadIdx.x & 31;
  int half = lane >> 4;
  int lrow = lane & 15;
  unsigned ldsA[2] = { (unsigned)(size_t)&shA[wv][0][0],
                       (unsigned)(size_t)&shA[wv][1][0] };
  unsigned ldsB[2] = { (unsigned)(size_t)&shB[wv][0][0],
                       (unsigned)(size_t)&shB[wv][1][0] };

  // prologue: stage k0 = 0 into buffer 0
  stage_panel(Ab, m0, lda, 0, lane, ldsA[0]);
  stage_panel(Bb, n0, ldb, 0, lane, ldsB[0]);

  v8f acc = {0.f,0.f,0.f,0.f,0.f,0.f,0.f,0.f};
  int buf = 0;
  for (int k0 = 0; k0 < K; k0 += 32, buf ^= 1) {
    bool more = (k0 + 32 < K);               // uniform
    if (more) {
      stage_panel(Ab, m0, lda, k0 + 32, lane, ldsA[buf^1]);
      stage_panel(Bb, n0, ldb, k0 + 32, lane, ldsB[buf^1]);
      wait_async8();                         // <=8 outstanding => step k0 landed
    } else {
      wait_async0();
    }
    const float* sAp = &shA[wv][buf][0];
    const float* sBp = &shB[wv][buf][0];
    v16h a, b;
#pragma unroll
    for (int j = 0; j < 8; ++j) {
      int ka = ((j < 4) ? (2*j) : (2*j + 8)) + 8*half;     // A frag K index
      float2 av = *(const float2*)&sAp[lrow*LDS_RS + ka];
      a[2*j] = (f16)av.x; a[2*j+1] = (f16)av.y;
      int kb = ((j < 4) ? (2*j) : (2*j + 8)) + 8*half;     // B frag K index
      float2 bv = *(const float2*)&sBp[lrow*LDS_RS + kb];
      b[2*j] = (f16)bv.x; b[2*j+1] = (f16)bv.y;
    }
    acc = __builtin_amdgcn_wmma_f32_16x16x32_f16(false, a, false, b, (short)0, acc,
                                                 false, false);
  }
  int bn = n0 + lrow;
  float bi  = bias  ? bias[bn]       : 0.f;
  float scv = scale ? scale[bn]*sfac : 1.f;
  float shv = scale ? shift[bn]      : 0.f;
  const float* resp = HASRES ? (res + (long)bz*sR + bn) : nullptr;
#pragma unroll
  for (int j = 0; j < 8; ++j) {
    int m = m0 + j + 8*half;
    float v = acc[j] + bi;
    if (scale) v = v*scv + shv;
    if (ACT == 1) v = geluf(v);
    else if (ACT == 2) v = siluf(v);
    if (HASRES) v += resp[(long)m*ldc];
    Cb[(long)m*ldc + bn] = v;
  }
}

// ---------------------------------------------------------------------------
// Guarded WMMA GEMM for ragged shapes (x_proj M=38/44, dt-mix K=6/12)
// ---------------------------------------------------------------------------
template<int TRANSB, int ACT>
__global__ void wmma_gemm_kernel(
    const float* __restrict__ A, const float* __restrict__ Bm, float* __restrict__ C,
    const float* __restrict__ bias, const float* __restrict__ scale,
    const float* __restrict__ shift, const float* __restrict__ res,
    int M, int N, int K, int lda, int ldb, int ldc,
    long sA, long sB, long sC, long sR, int amod, float sfac)
{
  int wid = blockIdx.x * (blockDim.x >> 5) + (threadIdx.x >> 5);
  int tnc = (N + 15) >> 4;
  int tiles = ((M + 15) >> 4) * tnc;
  if (wid >= tiles) return;                 // wave-uniform
  int bz = blockIdx.z;
  const float* Ab = A + (amod > 0 ? (long)(bz % amod) : (long)bz) * sA;
  const float* Bb = Bm + (long)bz * sB;
  float* Cb = C + (long)bz * sC;
  int tm = wid / tnc, tn = wid % tnc;
  int m0 = tm << 4, n0 = tn << 4;
  int lane = threadIdx.x & 31;
  int half = lane >> 4;
  int lrow = lane & 15;
  int am = m0 + lrow;
  int bn = n0 + lrow;
  v8f acc = {0.f,0.f,0.f,0.f,0.f,0.f,0.f,0.f};
  for (int k0 = 0; k0 < K; k0 += 32) {
    v16h a, b;
#pragma unroll
    for (int j = 0; j < 8; ++j) {
      int ka = k0 + ((j < 4) ? (2*j) : (2*j + 8)) + 8*half;
      float a0 = 0.f, a1 = 0.f;
      if (am < M) {
        if (ka < K)     a0 = Ab[(long)am*lda + ka];
        if (ka + 1 < K) a1 = Ab[(long)am*lda + ka + 1];
      }
      a[2*j] = (f16)a0; a[2*j+1] = (f16)a1;
      int kb = k0 + ((j < 4) ? (2*j) : (2*j + 8)) + 8*half;
      float b0 = 0.f, b1 = 0.f;
      if (bn < N) {
        if (TRANSB) {
          if (kb < K)     b0 = Bb[(long)bn*ldb + kb];
          if (kb + 1 < K) b1 = Bb[(long)bn*ldb + kb + 1];
        } else {
          if (kb < K)     b0 = Bb[(long)kb*ldb + bn];
          if (kb + 1 < K) b1 = Bb[(long)(kb+1)*ldb + bn];
        }
      }
      b[2*j] = (f16)b0; b[2*j+1] = (f16)b1;
    }
    acc = __builtin_amdgcn_wmma_f32_16x16x32_f16(false, a, false, b, (short)0, acc,
                                                 false, false);
  }
  if (bn < N) {
    float bi  = bias  ? bias[bn]       : 0.f;
    float scv = scale ? scale[bn]*sfac : 1.f;
    float shv = scale ? shift[bn]      : 0.f;
#pragma unroll
    for (int j = 0; j < 8; ++j) {
      int m = m0 + j + 8*half;
      if (m < M) {
        float v = acc[j] + bi;
        if (scale) v = v*scv + shv;
        if (ACT == 1) v = geluf(v);
        else if (ACT == 2) v = siluf(v);
        if (res) v += res[(long)bz*sR + (long)m*ldc + bn];
        Cb[(long)m*ldc + bn] = v;
      }
    }
  }
}

static void gemm(hipStream_t st, int transb, int act,
                 const float* A, const float* B, float* C,
                 int M, int N, int K, int lda, int ldb, int ldc,
                 long sA, long sB, long sC, int amod, int batches,
                 const float* bias, const float* scale, const float* shift, float sfac,
                 const float* res, long sR)
{
  bool fast = transb && (M % 16 == 0) && (N % 16 == 0) && (K % 32 == 0);
  if (fast) {
    int tiles = (M/16)*(N/16);
    dim3 grid((unsigned)((tiles + 7)/8), 1, (unsigned)batches), block(256,1,1);
#define GOF(AC,RS) wmma_gemm_fast_kernel<AC,RS><<<grid,block,0,st>>>(A,B,C,bias,scale,shift,res,\
                   M,N,K,lda,ldb,ldc,sA,sB,sC,sR,amod,sfac)
    if (act == 1) { if (res) GOF(1,1); else GOF(1,0); }
    else          { if (res) GOF(0,1); else GOF(0,0); }
#undef GOF
  } else {
    int tiles = ((M+15)/16)*((N+15)/16);
    dim3 grid((unsigned)((tiles + 3)/4), 1, (unsigned)batches), block(128,1,1);
#define GO(TB,AC) wmma_gemm_kernel<TB,AC><<<grid,block,0,st>>>(A,B,C,bias,scale,shift,res,\
                  M,N,K,lda,ldb,ldc,sA,sB,sC,sR,amod,sfac)
    if (transb) { if (act==1) GO(1,1); else if (act==2) GO(1,2); else GO(1,0); }
    else        { GO(0,0); }
#undef GO
  }
}

// ---------------------------------------------------------------------------
// Elementwise / reduction kernels
// ---------------------------------------------------------------------------
__global__ void patch_embed_kernel(const float* __restrict__ x, const float* __restrict__ w,
                                   const float* __restrict__ b, float* __restrict__ out,
                                   int B, int Co, int Hp, int Wp)
{
  long tid = (long)blockIdx.x*blockDim.x + threadIdx.x;
  long total = (long)B*Hp*Wp*Co;
  if (tid >= total) return;
  int c = (int)(tid % Co); long t = tid / Co;
  int ww = (int)(t % Wp); int hh = (int)((t / Wp) % Hp); int bb = (int)(t / ((long)Wp*Hp));
  const int Hi = Hp*4, Wi = Wp*4;
  float acc = b[c];
  for (int ci = 0; ci < 3; ++ci)
    for (int kh = 0; kh < 4; ++kh)
      for (int kw = 0; kw < 4; ++kw)
        acc += x[(((long)bb*3 + ci)*Hi + hh*4 + kh)*Wi + ww*4 + kw]
             * w[((c*3 + ci)*4 + kh)*4 + kw];
  out[tid] = acc;
}

__global__ void ln_kernel(const float* __restrict__ in, const float* __restrict__ g,
                          const float* __restrict__ bta, float* __restrict__ out,
                          long T, int Cn)
{
  long wid = (long)blockIdx.x*(blockDim.x >> 5) + (threadIdx.x >> 5);
  if (wid >= T) return;
  int lane = threadIdx.x & 31;
  const float* r = in + wid*Cn;
  float s = 0.f, ss = 0.f;
  for (int c = lane; c < Cn; c += 32) { float v = r[c]; s += v; ss += v*v; }
  s = wave_sum(s); ss = wave_sum(ss);
  float m = s / Cn;
  float var = ss / Cn - m*m;
  float rs = rsqrtf(var + 1e-5f);
  float* o = out + wid*Cn;
  for (int c = lane; c < Cn; c += 32) o[c] = (r[c]-m)*rs*g[c] + bta[c];
}

// forward real DFT along W (NHWC token input), ortho per-axis scale 1/sqrt(W)
__global__ void dft_rows_kernel(const float* __restrict__ tok, float2* __restrict__ Xf,
                                int B, int C, int H, int W, int WF)
{
  long tid = (long)blockIdx.x*blockDim.x + threadIdx.x;
  long total = (long)B*C*H*WF;
  if (tid >= total) return;
  int kf = (int)(tid % WF); long t = tid / WF;
  int h = (int)(t % H); t /= H;
  int c = (int)(t % C); int b = (int)(t / C);
  float w0 = -2.f*PI_F*(float)kf/(float)W;
  float sr = 0.f, si = 0.f;
  for (int w = 0; w < W; ++w) {
    float v = tok[(((long)(b*H + h)*W) + w)*C + c];
    float sn, cs; __sincosf(w0*(float)w, &sn, &cs);
    sr += v*cs; si += v*sn;
  }
  float sc = rsqrtf((float)W);
  Xf[tid] = make_float2(sr*sc, si*sc);
}

// complex DFT along H (sign=-1 fwd, +1 inv), ortho per-axis scale 1/sqrt(H)
__global__ void dft_cols_kernel(const float2* __restrict__ in, float2* __restrict__ out,
                                int B, int C, int H, int WF, float sign)
{
  long tid = (long)blockIdx.x*blockDim.x + threadIdx.x;
  long total = (long)B*C*H*WF;
  if (tid >= total) return;
  int wf = (int)(tid % WF); long t = tid / WF;
  int kh = (int)(t % H); long bc = t / H;
  float w0 = sign*2.f*PI_F*(float)kh/(float)H;
  float sr = 0.f, si = 0.f;
  for (int h = 0; h < H; ++h) {
    float2 z = in[(bc*H + h)*WF + wf];
    float sn, cs; __sincosf(w0*(float)h, &sn, &cs);
    sr += z.x*cs - z.y*sn;
    si += z.x*sn + z.y*cs;
  }
  float sc = rsqrtf((float)H);
  out[(bc*H + kh)*WF + wf] = make_float2(sr*sc, si*sc);
}

// FGE gating: low/high masks partition the spectrum -> branch(z)+branch(0)
__global__ void fge_gate_kernel(const float2* __restrict__ Xf, const float* __restrict__ w1,
                                const float* __restrict__ b1, float2* __restrict__ out,
                                int B, int Cc, int H, int WF)
{
  long tid = (long)blockIdx.x*blockDim.x + threadIdx.x;
  long total = (long)B*4*24*H*WF;
  if (tid >= total) return;
  int wf = (int)(tid % WF); long t = tid / WF;
  int h = (int)(t % H); t /= H;
  int o = (int)(t % 24); t /= 24;
  int g = (int)(t % 4); int b = (int)(t / 4);
  long pix = (long)h*WF + wf;
  long chanBase = (long)b*Cc + g*24;
  float sr = 0.f, si = 0.f;
  for (int i = 0; i < 24; ++i) {
    float2 z = Xf[(chanBase + i)*(long)H*WF + pix];
    sr += z.x * w1[((0*4 + g)*24 + i)*24 + o];
    si += z.y * w1[((4   + g)*24 + i)*24 + o];
  }
  float2 zo = Xf[(chanBase + o)*(long)H*WF + pix];
  float br = b1[(0*4 + g)*24 + o];
  float bi = b1[(4   + g)*24 + o];
  float outr = geluf(sr + br) + zo.x + geluf(br);   // data-branch + zero-branch
  float outi = geluf(si + bi) + zo.y + geluf(bi);
  outr = sshrink(outr) + zo.x;                      // softshrink then + orig
  outi = sshrink(outi) + zo.y;
  out[(chanBase + o)*(long)H*WF + pix] = make_float2(outr, outi);
}

// hermitian inverse DFT along W -> real NHWC tokens
__global__ void idft_rows_kernel(const float2* __restrict__ in, float* __restrict__ tok,
                                 int B, int C, int H, int W, int WF)
{
  long tid = (long)blockIdx.x*blockDim.x + threadIdx.x;
  long total = (long)B*C*H*W;
  if (tid >= total) return;
  int w = (int)(tid % W); long t = tid / W;
  int h = (int)(t % H); t /= H;
  int c = (int)(t % C); int b = (int)(t / C);
  float w0 = 2.f*PI_F*(float)w/(float)W;
  float acc = 0.f;
  for (int k = 0; k < WF; ++k) {
    float2 z = in[(((long)(b*C + c)*H) + h)*WF + k];
    float sn, cs; __sincosf(w0*(float)k, &sn, &cs);
    float wt = (k == 0 || 2*k == W) ? 1.f : 2.f;
    acc += wt*(z.x*cs - z.y*sn);
  }
  tok[(((long)(b*H + h)*W) + w)*C + c] = acc * rsqrtf((float)W);
}

// depthwise 3x3 + silu; input xc = first di channels of xz tokens; out (B,di,L)
__global__ void dwconv_silu_kernel(const float* __restrict__ xz, const float* __restrict__ cw,
                                   const float* __restrict__ cb, float* __restrict__ out,
                                   int B, int di, int H, int W, int ld)
{
  long tid = (long)blockIdx.x*blockDim.x + threadIdx.x;
  long total = (long)B*di*H*W;
  if (tid >= total) return;
  int w = (int)(tid % W); long t = tid / W;
  int h = (int)(t % H); t /= H;
  int c = (int)(t % di); int b = (int)(t / di);
  float acc = cb[c];
  for (int kh = 0; kh < 3; ++kh) {
    int ih = h + kh - 1;
    if (ih < 0 || ih >= H) continue;
    for (int kw = 0; kw < 3; ++kw) {
      int iw = w + kw - 1;
      if (iw < 0 || iw >= W) continue;
      acc += xz[(((long)(b*H + ih)*W) + iw)*ld + c] * cw[c*9 + kh*3 + kw];
    }
  }
  out[(((long)b*di + c)*(long)H*W) + (long)h*W + w] = siluf(acc);
}

// build 4 scan directions
__global__ void build_xs_kernel(const float* __restrict__ xc, float* __restrict__ xs,
                                int B, int di, int H, int W)
{
  long L = (long)H*W;
  long tid = (long)blockIdx.x*blockDim.x + threadIdx.x;
  long total = (long)B*di*L;
  if (tid >= total) return;
  long l = tid % L; long t = tid / L;
  int d = (int)(t % di); int b = (int)(t / di);
  const float* inp = xc + ((long)b*di + d)*L;
  float vhw = inp[l];
  int h2 = (int)(l % H); long w2 = l / H;
  float vwh = inp[(long)h2*W + w2];
  xs[(((long)b*4 + 0)*di + d)*L + l]           = vhw;
  xs[(((long)b*4 + 1)*di + d)*L + l]           = vwh;
  xs[(((long)b*4 + 2)*di + d)*L + (L - 1 - l)] = vhw;
  xs[(((long)b*4 + 3)*di + d)*L + (L - 1 - l)] = vwh;
}

// selective scan: one thread per (b,k,d), 16-wide state in VGPRs
__global__ void scan_kernel(const float* __restrict__ xs, const float* __restrict__ dts,
                            const float* __restrict__ A_log, const float* __restrict__ xdbl,
                            const float* __restrict__ Ds, const float* __restrict__ dt_b,
                            float* __restrict__ ys, int B, int di, int L, int Crows, int R)
{
  const int N = 16;
  int tid = blockIdx.x*blockDim.x + threadIdx.x;
  int total = B*4*di;
  if (tid >= total) return;
  int d = tid % di; int k = (tid/di) % 4; int b = tid/(4*di);
  float Av[N], h[N];
#pragma unroll
  for (int n = 0; n < N; ++n) { Av[n] = -__expf(A_log[((long)k*di + d)*N + n]); h[n] = 0.f; }
  long xoff = ((long)(b*4 + k)*di + d)*L;
  long doff = (long)(b*4 + k)*Crows*L;
  float dbias = dt_b[k*di + d];
  float Dv = Ds[k*di + d];
  for (int l = 0; l < L; ++l) {
    float u  = xs[xoff + l];
    float dt = softplusf(dts[xoff + l] + dbias);
    float du = dt*u;
    float y = 0.f;
#pragma unroll
    for (int n = 0; n < N; ++n) {
      float bn = xdbl[doff + (long)(R + n)*L + l];
      float cn = xdbl[doff + (long)(R + N + n)*L + l];
      h[n] = h[n]*__expf(dt*Av[n]) + du*bn;
      y += h[n]*cn;
    }
    ys[xoff + l] = y + Dv*u;
  }
}

// merge the 4 directional outputs into token-major (B*L, di)
__global__ void combine_y_kernel(const float* __restrict__ ys, float* __restrict__ out,
                                 int B, int di, int H, int W)
{
  long L = (long)H*W;
  long tid = (long)blockIdx.x*blockDim.x + threadIdx.x;
  long total = (long)B*L*di;
  if (tid >= total) return;
  int d = (int)(tid % di); long t = tid / di;
  long l = t % L; int b = (int)(t / L);
  int w = (int)(l % W); int h = (int)(l / W);
  long bb = (long)b*4;
  long lt = (long)w*H + h;
  float y1 = ys[((bb + 0)*di + d)*L + l];
  float y2 = ys[((bb + 2)*di + d)*L + (L - 1 - l)];
  float y3 = ys[((bb + 1)*di + d)*L + lt];
  float y4 = ys[((bb + 3)*di + d)*L + (L - 1 - lt)];
  out[tid] = y1 + y2 + y3 + y4;
}

__global__ void gate_mul_kernel(float* __restrict__ y, const float* __restrict__ xz,
                                long T, int di, int ld)
{
  long tid = (long)blockIdx.x*blockDim.x + threadIdx.x;
  if (tid >= T*di) return;
  int d = (int)(tid % di); long t = tid / di;
  y[tid] *= siluf(xz[t*ld + di + d]);
}

__global__ void merge_kernel(const float* __restrict__ in, float* __restrict__ out,
                             int B, int H, int W, int Cn)
{
  int H2 = H/2, W2 = W/2;
  long tid = (long)blockIdx.x*blockDim.x + threadIdx.x;
  long total = (long)B*H2*W2*4*Cn;
  if (tid >= total) return;
  int c = (int)(tid % Cn); long t = tid / Cn;
  int q = (int)(t % 4); t /= 4;
  int j = (int)(t % W2); t /= W2;
  int i = (int)(t % H2); int b = (int)(t / H2);
  int dh = q & 1, dw = q >> 1;
  out[tid] = in[(((long)(b*H) + 2*i + dh)*W + 2*j + dw)*Cn + c];
}

__global__ void pool_kernel(const float* __restrict__ in, float* __restrict__ out,
                            int B, long T, int Cn)
{
  int tid = blockIdx.x*blockDim.x + threadIdx.x;
  if (tid >= B*Cn) return;
  int c = tid % Cn; int b = tid / Cn;
  float acc = 0.f;
  for (long t = 0; t < T; ++t) acc += in[((long)b*T + t)*Cn + c];
  out[tid] = acc / (float)T;
}

__global__ void cls_kernel(const float* __restrict__ pooled, const float* __restrict__ w,
                           const float* __restrict__ b, float* __restrict__ out,
                           int B, int Cn, int O)
{
  int tid = blockIdx.x*blockDim.x + threadIdx.x;
  if (tid >= B*O) return;
  int o = tid % O; int bb = tid / O;
  float acc = b[o];
  for (int c = 0; c < Cn; ++c) acc += pooled[bb*Cn + c]*w[o*Cn + c];
  out[tid] = acc;
}

// ---------------------------------------------------------------------------
// Host orchestration
// ---------------------------------------------------------------------------
struct Bump {
  char* base; size_t off;
  float* take(long n){
    float* p = (float*)(base + off);
    off += (size_t)n*sizeof(float);
    off = (off + 255) & ~(size_t)255;
    return p;
  }
};

struct SS2DParams {
  const float *ln_g, *ln_b, *in_proj, *conv_w, *conv_b, *x_proj, *dt_w,
              *dt_b, *A_log, *Ds, *on_g, *on_b, *out_proj;
};

static void run_ss2d(hipStream_t st, const float* y, float* ynew, int B, int H, int W,
                     int d, const SS2DParams& p, Bump b)
{
  const int N = 16;
  int di = 2*d;
  int R = (d + 15)/16;
  int Crows = R + 2*N;
  long L = (long)H*W;
  long T = (long)B*L;

  float* lnb  = b.take(T*d);
  float* xz   = b.take(T*2*di);
  float* xc   = b.take((long)B*di*L);
  float* xs   = b.take((long)B*4*di*L);
  float* xdbl = b.take((long)B*4*Crows*L);
  float* dts  = b.take((long)B*4*di*L);
  float* ysb  = b.take((long)B*4*di*L);
  float* yc   = b.take(T*di);
  float* ln2  = b.take(T*di);

  ln_kernel<<<cdiv_h(T,8),256,0,st>>>(y, p.ln_g, p.ln_b, lnb, T, d);
  // xz = ln @ in_proj.T
  gemm(st, 1, 0, lnb, p.in_proj, xz, (int)T, 2*di, d, d, d, 2*di,
       0,0,0, 0, 1, nullptr,nullptr,nullptr,1.f, nullptr,0);
  dwconv_silu_kernel<<<cdiv_h((long)B*di*L,256),256,0,st>>>(xz, p.conv_w, p.conv_b, xc,
                                                            B, di, H, W, 2*di);
  build_xs_kernel<<<cdiv_h((long)B*di*L,256),256,0,st>>>(xc, xs, B, di, H, W);
  // x_dbl[b,k] = x_proj[k] (Crows x di) @ xs[b,k] (di x L)
  gemm(st, 0, 0, p.x_proj, xs, xdbl, Crows, (int)L, di, di, (int)L, (int)L,
       (long)Crows*di, (long)di*L, (long)Crows*L, 4, B*4,
       nullptr,nullptr,nullptr,1.f, nullptr,0);
  // dts[b,k] = dt_w[k] (di x R) @ x_dbl[b,k][0:R] (R x L)
  gemm(st, 0, 0, p.dt_w, xdbl, dts, di, (int)L, R, R, (int)L, (int)L,
       (long)di*R, (long)Crows*L, (long)di*L, 4, B*4,
       nullptr,nullptr,nullptr,1.f, nullptr,0);
  scan_kernel<<<cdiv_h(B*4*di,128),128,0,st>>>(xs, dts, p.A_log, xdbl, p.Ds, p.dt_b,
                                               ysb, B, di, (int)L, Crows, R);
  combine_y_kernel<<<cdiv_h(T*di,256),256,0,st>>>(ysb, yc, B, di, H, W);
  ln_kernel<<<cdiv_h(T,8),256,0,st>>>(yc, p.on_g, p.on_b, ln2, T, di);
  gate_mul_kernel<<<cdiv_h(T*di,256),256,0,st>>>(ln2, xz, T, di, 2*di);
  // ynew = y + ln2 @ out_proj.T
  gemm(st, 1, 0, ln2, p.out_proj, ynew, (int)T, d, di, di, di, d,
       0,0,0, 0, 1, nullptr,nullptr,nullptr,1.f, y, 0);
}

extern "C" void kernel_launch(void* const* d_in, const int* in_sizes, int n_in,
                              void* d_out, int out_size, void* d_ws, size_t ws_size,
                              hipStream_t stream)
{
  (void)in_sizes; (void)n_in; (void)out_size; (void)ws_size;
  int pi = 0;
  auto F = [&]() -> const float* { return (const float*)d_in[pi++]; };

  const float* x        = F();
  const float* pe_w     = F();
  const float* pe_b     = F();
  const float* pe_ln_g  = F();
  const float* pe_ln_b  = F();
  const float* fge_w1   = F();
  const float* fge_b1   = F();
  const float* fge_pw   = F();
  const float* fge_pb   = F();
  const float* fge_bn_g = F();
  const float* fge_bn_b = F();
  SS2DParams s1 { F(),F(),F(),F(),F(),F(),F(),F(),F(),F(),F(),F(),F() };
  const float* pm_ln_g  = F();
  const float* pm_ln_b  = F();
  const float* pm_w     = F();
  SS2DParams s2 { F(),F(),F(),F(),F(),F(),F(),F(),F(),F(),F(),F(),F() };
  const float* cls_w    = F();
  const float* cls_b    = F();

  const int B = 2;
  const float bnfac = 0.99999500003750f;   // 1/sqrt(1+1e-5)

  Bump ws { (char*)d_ws, 0 };
  float* yA     = ws.take((long)8192*96);
  float* yB     = ws.take((long)8192*96);
  float* y2a    = ws.take((long)2048*192);
  float* y2b    = ws.take((long)2048*192);
  float* pooled = ws.take(512);
  Bump scratch = ws;   // everything below here is phase-local and reused

  // 1) patch embed (NCHW conv4x4/s4 -> NHWC tokens)
  patch_embed_kernel<<<cdiv_h((long)8192*96,256),256,0,stream>>>(x, pe_w, pe_b, yA,
                                                                 B, 96, 64, 64);
  // 2) pe layernorm
  ln_kernel<<<cdiv_h(8192,8),256,0,stream>>>(yA, pe_ln_g, pe_ln_b, yB, 8192, 96);

  // 3) FGE block: rfft2 -> gate -> irfft2 -> proj(+bn)+gelu, result into yA
  {
    Bump b = scratch;
    const int C = 96, H = 64, W = 64, WF = 33;
    float2* Xr = (float2*)b.take((long)B*C*H*WF*2);
    float2* Xc = (float2*)b.take((long)B*C*H*WF*2);
    float2* Xg = (float2*)b.take((long)B*C*H*WF*2);
    float2* Xi = (float2*)b.take((long)B*C*H*WF*2);
    float*  fsp = b.take((long)B*H*W*C);
    long tf = (long)B*C*H*WF;
    dft_rows_kernel<<<cdiv_h(tf,256),256,0,stream>>>(yB, Xr, B, C, H, W, WF);
    dft_cols_kernel<<<cdiv_h(tf,256),256,0,stream>>>(Xr, Xc, B, C, H, WF, -1.f);
    long tg = (long)B*4*24*H*WF;
    fge_gate_kernel<<<cdiv_h(tg,256),256,0,stream>>>(Xc, fge_w1, fge_b1, Xg, B, C, H, WF);
    dft_cols_kernel<<<cdiv_h(tf,256),256,0,stream>>>(Xg, Xi, B, C, H, WF, 1.f);
    long ti = (long)B*C*H*W;
    idft_rows_kernel<<<cdiv_h(ti,256),256,0,stream>>>(Xi, fsp, B, C, H, W, WF);
    // gelu((x@proj_w.T + proj_b)*bn_g/sqrt(1+eps) + bn_b)
    gemm(stream, 1, 1, fsp, fge_pw, yA, 8192, 96, 96, 96, 96, 96,
         0,0,0, 0, 1, fge_pb, fge_bn_g, fge_bn_b, bnfac, nullptr, 0);
  }

  // 4) stage1 SS2D (d=96, 64x64), residual inside -> yB
  run_ss2d(stream, yA, yB, B, 64, 64, 96, s1, scratch);

  // 5) patch merge + LN + pm projection -> y2a
  {
    Bump b = scratch;
    float* merged = b.take((long)2048*384);
    float* mln    = b.take((long)2048*384);
    merge_kernel<<<cdiv_h((long)2048*384,256),256,0,stream>>>(yB, merged, B, 64, 64, 96);
    ln_kernel<<<cdiv_h(2048,8),256,0,stream>>>(merged, pm_ln_g, pm_ln_b, mln, 2048, 384);
    gemm(stream, 1, 0, mln, pm_w, y2a, 2048, 192, 384, 384, 384, 192,
         0,0,0, 0, 1, nullptr,nullptr,nullptr,1.f, nullptr,0);
  }

  // 6) stage2 SS2D (d=192, 32x32) -> y2b
  run_ss2d(stream, y2a, y2b, B, 32, 32, 192, s2, scratch);

  // 7) global mean pool + classifier
  pool_kernel<<<cdiv_h(B*192,128),128,0,stream>>>(y2b, pooled, B, 1024, 192);
  cls_kernel<<<cdiv_h(B*1000,128),128,0,stream>>>(pooled, cls_w, cls_b,
                                                  (float*)d_out, B, 192, 1000);
}
